// LocalEnhancementModule_48576080118247
// MI455X (gfx1250) — matrix operand order repaired and checked by reading, not compiled, not run on hardware
//
#include <hip/hip_runtime.h>
#include <hip/hip_bf16.h>
#include <cstdint>

// ---------------------------------------------------------------------------
// LocalEnhancementModule for MI455X (gfx1250, wave32, WMMA)
// B=16, C=16, H=W=256, NP=16 -> P=256, D=4096, E=512
// GEMMs: v_wmma_f32_16x16x32_bf16, tiles staged via
// global_load_async_to_lds_b128 (ASYNCcnt) with double-buffered LDS.
// ---------------------------------------------------------------------------

typedef __attribute__((ext_vector_type(16))) __bf16 v16bf;
typedef __attribute__((ext_vector_type(8)))  __bf16 v8bf;
typedef __attribute__((ext_vector_type(4)))  __bf16 v4bf;
typedef __attribute__((ext_vector_type(8)))  float  v8f;

struct bf16x16 { v8bf lo, hi; };

#define B_  16
#define C_  16
#define H_  256
#define W_  256
#define NP_ 16
#define P_  256      // NP*NP
#define D_  4096     // C*16*16
#define E_  512

#define BK_   32     // K-tile depth (one WMMA K step)
#define LDK_  40     // padded LDS row stride in bf16 (80B: 16B-aligned, conflict-free)

// ---------------------------------------------------------------------------
// 1) Patch extraction + f32 -> bf16:  x[B,C,H,W] -> p[B*P, D]
// ---------------------------------------------------------------------------
__global__ __launch_bounds__(256) void patchify_bf16(
    const float* __restrict__ x, __bf16* __restrict__ p)
{
    long idx4 = (long)blockIdx.x * 256 + threadIdx.x;   // B*C*H*W/4 threads
    long base = idx4 * 4;
    int w = (int)(base & 255);
    int h = (int)((base >> 8) & 255);
    int c = (int)((base >> 16) & 15);
    int b = (int)(base >> 20);
    int ix = w & 15, px = w >> 4;
    int iy = h & 15, py = h >> 4;
    int pidx = py * NP_ + px;
    int n = iy * 16 + ix;                                // n % 4 == 0
    float4 xv = *(const float4*)(x + base);
    v4bf o;
    o[0] = (__bf16)xv.x; o[1] = (__bf16)xv.y;
    o[2] = (__bf16)xv.z; o[3] = (__bf16)xv.w;
    long dst = ((long)(b * P_ + pidx)) * D_ + c * 256 + n;
    *(v4bf*)(p + dst) = o;
}

// ---------------------------------------------------------------------------
// 2) Tiled transpose + f32 -> bf16:  src[R,Ccols] f32 -> dst[Ccols,R] bf16
//    (weights become [N,K] row-major so GEMM staging is fully contiguous)
// ---------------------------------------------------------------------------
__global__ __launch_bounds__(256) void transpose_cvt_bf16(
    const float* __restrict__ src, __bf16* __restrict__ dst, int R, int Ccols)
{
    __shared__ float tile[32][33];
    int c0 = blockIdx.x * 32;
    int r0 = blockIdx.y * 32;
    int tx = threadIdx.x & 31;
    int ty = threadIdx.x >> 5;          // 0..7
    #pragma unroll
    for (int i = 0; i < 32; i += 8)
        tile[ty + i][tx] = src[(long)(r0 + ty + i) * Ccols + c0 + tx];
    __syncthreads();
    #pragma unroll
    for (int i = 0; i < 32; i += 8)
        dst[(long)(c0 + ty + i) * R + r0 + tx] = (__bf16)tile[tx][ty + i];
}

// ---------------------------------------------------------------------------
// 3) Tiled bf16 WMMA GEMM:  C[M,N] = A[M,K] @ B[N,K]^T + bias
//    A [M,K] row-major, B [N,K] row-major (pre-transposed weights / f-matrix).
//    BM=BN=128, BK=32; 256 threads = 8 waves; wave tile = 32(M) x 64(N).
//    Tiles staged with global_load_async_to_lds_b128, double-buffered LDS.
//    M,N multiples of 128; K multiple of 32 (true for all uses here).
// ---------------------------------------------------------------------------
template <bool OUT_BF16>
__global__ __launch_bounds__(256) void gemm_wmma_bf16(
    const __bf16* __restrict__ A, const __bf16* __restrict__ Bm,
    const float* __restrict__ bias, void* __restrict__ Cout,
    int M, int N, int K, long strideA, long strideB, long strideC)
{
    // [buf][row][k] ; row stride 40 bf16 = 80B (16B aligned, conflict-free)
    __shared__ __align__(16) __bf16 As[2][128][LDK_];   // 2 x 10240 B
    __shared__ __align__(16) __bf16 Bs[2][128][LDK_];   // 2 x 10240 B

    const int tid   = threadIdx.x;
    const int wave  = tid >> 5;
    const int lane  = tid & 31;
    const int waveM = wave >> 1;          // 0..3  -> M offset waveM*32
    const int waveN = wave & 1;           // 0..1  -> N offset waveN*64
    const int r16   = lane & 15;
    const int half  = lane >> 4;          // 0/1 lane half
    const int kA    = half * 8;           // A frag K split {0-7,16-23}/{8-15,24-31}
    const int kB    = half * 8;           // B frag K split {0-7,16-23}/{8-15,24-31}

    const long batch = blockIdx.z;
    const int blockM = blockIdx.y * 128;
    const int blockN = blockIdx.x * 128;
    A  += batch * strideA + (long)blockM * K;
    Bm += batch * strideB + (long)blockN * K;

    // per-thread staging geometry: 2 x 16B chunks per matrix per tile
    int      rowc[2], kcc[2];
    unsigned ldsA[2], ldsB[2];
    #pragma unroll
    for (int i = 0; i < 2; ++i) {
        int cid = tid + i * 256;          // 0..511
        rowc[i] = cid >> 2;               // 0..127
        kcc[i]  = (cid & 3) * 8;          // 0,8,16,24
        // low 32 bits of a flat LDS pointer == wave-relative LDS address
        ldsA[i] = (unsigned)(uintptr_t)&As[0][rowc[i]][kcc[i]];
        ldsB[i] = (unsigned)(uintptr_t)&Bs[0][rowc[i]][kcc[i]];
    }
    const unsigned bufStep = (unsigned)(sizeof(__bf16) * 128 * LDK_);  // 10240

    const int ntiles = K / BK_;

    auto issue_tile = [&](int kt, int buf) {
        const unsigned add = buf ? bufStep : 0u;
        #pragma unroll
        for (int i = 0; i < 2; ++i) {
            uint64_t ga = (uint64_t)(uintptr_t)(A + (long)rowc[i] * K + kt * BK_ + kcc[i]);
            asm volatile("global_load_async_to_lds_b128 %0, %1, off"
                         :: "v"(ldsA[i] + add), "v"(ga) : "memory");
        }
        #pragma unroll
        for (int i = 0; i < 2; ++i) {
            uint64_t gb = (uint64_t)(uintptr_t)(Bm + (long)rowc[i] * K + kt * BK_ + kcc[i]);
            asm volatile("global_load_async_to_lds_b128 %0, %1, off"
                         :: "v"(ldsB[i] + add), "v"(gb) : "memory");
        }
    };

    v8f acc[2][4] = {};

    issue_tile(0, 0);
    int buf = 0;
    for (int kt = 0; kt < ntiles; ++kt) {
        // our async writes for buf have landed; barrier => everyone's have
        asm volatile("s_wait_asynccnt 0x0" ::: "memory");
        __syncthreads();
        if (kt + 1 < ntiles) issue_tile(kt + 1, buf ^ 1);

        // ---- A fragments per ISA 16-bit A layout
        v16bf afr[2];
        #pragma unroll
        for (int mi = 0; mi < 2; ++mi) {
            int rowA = waveM * 32 + mi * 16 + r16;
            bf16x16 t;
            t.lo = *(const v8bf*)(&As[buf][rowA][kA]);        // K = kA..kA+7
            t.hi = *(const v8bf*)(&As[buf][rowA][16 + kA]);   // K = 16+kA..+7
            afr[mi] = __builtin_bit_cast(v16bf, t);
        }
        // ---- B fragments + 8 WMMAs
        #pragma unroll
        for (int ni = 0; ni < 4; ++ni) {
            int colB = waveN * 64 + ni * 16 + r16;
            bf16x16 tb;
            tb.lo = *(const v8bf*)(&Bs[buf][colB][kB]);        // K = kB..kB+7
            tb.hi = *(const v8bf*)(&Bs[buf][colB][16 + kB]);   // K = 16+kB..+7
            v16bf bfr = __builtin_bit_cast(v16bf, tb);
            #pragma unroll
            for (int mi = 0; mi < 2; ++mi) {
                acc[mi][ni] = __builtin_amdgcn_wmma_f32_16x16x32_bf16(
                    false, afr[mi], false, bfr, (short)0, acc[mi][ni],
                    false, false);
            }
        }
        buf ^= 1;
    }

    // ---- epilogue: lane r16 = col, half*8+r = row (C/D 16x16 f32 layout)
    #pragma unroll
    for (int mi = 0; mi < 2; ++mi) {
        #pragma unroll
        for (int ni = 0; ni < 4; ++ni) {
            int gn = blockN + waveN * 64 + ni * 16 + r16;
            float bv = bias ? bias[gn] : 0.0f;
            #pragma unroll
            for (int r = 0; r < 8; ++r) {
                int gm = blockM + waveM * 32 + mi * 16 + half * 8 + r;
                float v = acc[mi][ni][r] + bv;
                long off = batch * strideC + (long)gm * N + gn;
                if constexpr (OUT_BF16) ((__bf16*)Cout)[off] = (__bf16)v;
                else                    ((float*)Cout)[off]  = v;
            }
        }
    }
}

// ---------------------------------------------------------------------------
// 4) Row softmax over 256-wide rows, in place. One block per row.
// ---------------------------------------------------------------------------
__global__ __launch_bounds__(256) void softmax256(float* __restrict__ w)
{
    __shared__ float red[256];
    long row = blockIdx.x;
    int  t   = threadIdx.x;
    float v = w[row * 256 + t];
    red[t] = v;
    __syncthreads();
    #pragma unroll
    for (int s = 128; s > 0; s >>= 1) {
        if (t < s) red[t] = fmaxf(red[t], red[t + s]);
        __syncthreads();
    }
    float mx = red[0];
    __syncthreads();
    float e = __expf(v - mx);
    red[t] = e;
    __syncthreads();
    #pragma unroll
    for (int s = 128; s > 0; s >>= 1) {
        if (t < s) red[t] += red[t + s];
        __syncthreads();
    }
    w[row * 256 + t] = e / red[0];
}

// ---------------------------------------------------------------------------
// 5) Fused gate * weights, inverse patchify, residual:
//    out[b,c,h,w] = scale * weights[b,pidx,n] * g[b,pidx,c*256+n] + x[b,c,h,w]
// ---------------------------------------------------------------------------
__global__ __launch_bounds__(256) void gate_unpatch(
    const float* __restrict__ x, const float* __restrict__ g,
    const float* __restrict__ wts, const float* __restrict__ scale,
    float* __restrict__ out)
{
    long idx4 = (long)blockIdx.x * 256 + threadIdx.x;
    long base = idx4 * 4;
    int w = (int)(base & 255);
    int h = (int)((base >> 8) & 255);
    int c = (int)((base >> 16) & 15);
    int b = (int)(base >> 20);
    int ix = w & 15, px = w >> 4;
    int iy = h & 15, py = h >> 4;
    int pidx = py * NP_ + px;
    int n = iy * 16 + ix;                         // n % 4 == 0
    long grow = ((long)(b * P_ + pidx)) * D_ + c * 256 + n;
    long wrow = ((long)(b * P_ + pidx)) * P_ + n;
    float4 xv = *(const float4*)(x + base);
    float4 gv = *(const float4*)(g + grow);
    float4 wv = *(const float4*)(wts + wrow);
    float s = scale[0];
    float4 o;
    o.x = s * wv.x * gv.x + xv.x;
    o.y = s * wv.y * gv.y + xv.y;
    o.z = s * wv.z * gv.z + xv.z;
    o.w = s * wv.w * gv.w + xv.w;
    *(float4*)(out + base) = o;
}

// ---------------------------------------------------------------------------
extern "C" void kernel_launch(void* const* d_in, const int* in_sizes, int n_in,
                              void* d_out, int out_size, void* d_ws, size_t ws_size,
                              hipStream_t stream)
{
    const float* x       = (const float*)d_in[0];
    const float* theta_w = (const float*)d_in[1];
    const float* theta_b = (const float*)d_in[2];
    const float* f_w     = (const float*)d_in[3];
    const float* f_b     = (const float*)d_in[4];
    const float* g_w     = (const float*)d_in[5];
    const float* g_b     = (const float*)d_in[6];
    const float* scale   = (const float*)d_in[7];

    // workspace carve-up (~148 MB total)
    char* ws = (char*)d_ws;
    size_t off = 0;
    __bf16* p_bf  = (__bf16*)(ws + off); off += (size_t)B_ * P_ * D_ * 2;   // 33.5MB
    __bf16* twT   = (__bf16*)(ws + off); off += (size_t)D_ * E_ * 2;        //  4.2MB  [E,D]
    __bf16* fwT   = (__bf16*)(ws + off); off += (size_t)D_ * E_ * 2;        //  4.2MB  [E,D]
    __bf16* gwT   = (__bf16*)(ws + off); off += (size_t)D_ * D_ * 2;        // 33.5MB  [D,D]
    __bf16* th_bf = (__bf16*)(ws + off); off += (size_t)B_ * P_ * E_ * 2;   //  4.2MB
    __bf16* ff_bf = (__bf16*)(ws + off); off += (size_t)B_ * P_ * E_ * 2;   //  4.2MB
    float*  wts   = (float*) (ws + off); off += (size_t)B_ * P_ * P_ * 4;   //  4.2MB
    float*  gbuf  = (float*) (ws + off); off += (size_t)B_ * P_ * D_ * 4;   // 67.1MB
    (void)ws_size; (void)in_sizes; (void)n_in; (void)out_size;

    const long MPD = (long)B_ * P_;   // 4096 rows

    // 1) patchify x -> p (bf16)
    patchify_bf16<<<(B_ * C_ * H_ * W_) / (4 * 256), 256, 0, stream>>>(x, p_bf);

    // 2) transpose+convert weights: W[D,N] f32 -> W^T[N,D] bf16
    transpose_cvt_bf16<<<dim3(E_ / 32, D_ / 32), 256, 0, stream>>>(theta_w, twT, D_, E_);
    transpose_cvt_bf16<<<dim3(E_ / 32, D_ / 32), 256, 0, stream>>>(f_w,     fwT, D_, E_);
    transpose_cvt_bf16<<<dim3(D_ / 32, D_ / 32), 256, 0, stream>>>(g_w,     gwT, D_, D_);

    // 3) theta = p @ theta_w + theta_b   [4096,512] (bf16 out)
    gemm_wmma_bf16<true><<<dim3(E_ / 128, MPD / 128, 1), 256, 0, stream>>>(
        p_bf, twT, theta_b, th_bf, (int)MPD, E_, D_, 0, 0, 0);
    // 4) f = p @ f_w + f_b               [4096,512] (bf16 out)
    gemm_wmma_bf16<true><<<dim3(E_ / 128, MPD / 128, 1), 256, 0, stream>>>(
        p_bf, fwT, f_b, ff_bf, (int)MPD, E_, D_, 0, 0, 0);

    // 5) logits[b] = theta[b] @ f[b]^T   batched [256,256], K=512 (f32 out)
    gemm_wmma_bf16<false><<<dim3(P_ / 128, P_ / 128, B_), 256, 0, stream>>>(
        th_bf, ff_bf, nullptr, wts, P_, P_, E_,
        (long)P_ * E_, (long)P_ * E_, (long)P_ * P_);

    // 6) softmax rows (in place)
    softmax256<<<B_ * P_, 256, 0, stream>>>(wts);

    // 7) g = p @ g_w + g_b               [4096,4096] (f32 out)
    gemm_wmma_bf16<false><<<dim3(D_ / 128, MPD / 128, 1), 256, 0, stream>>>(
        p_bf, gwT, g_b, gbuf, (int)MPD, D_, D_, 0, 0, 0);

    // 8) gated combine + inverse patchify + residual
    gate_unpatch<<<(B_ * C_ * H_ * W_) / (4 * 256), 256, 0, stream>>>(
        x, gbuf, wts, scale, (float*)d_out);
}
